// HAKMEMEntangledInterferenceLayer_43155831390563
// MI455X (gfx1250) — compile-verified, matrix operand order checked by reading, not run on hardware
//
#include <hip/hip_runtime.h>
#include <hip/hip_bf16.h>
#include <math.h>

#define S_LEN 2048
#define DIM   1024
#define HEADS 16
#define HD    64
#define ROT   32

typedef __attribute__((ext_vector_type(16))) _Float16     v16h;
typedef __attribute__((ext_vector_type(8)))  _Float16     v8h;
typedef __attribute__((ext_vector_type(8)))  float        v8f;
typedef __attribute__((ext_vector_type(4)))  unsigned int u32x4;
typedef __attribute__((ext_vector_type(8)))  int          i32x8;
typedef __attribute__((ext_vector_type(4)))  int          i32x4;

// Build a 16-element f16 fragment from two contiguous 8-element (16B) chunks.
// CDNA5 16-bit A/B fragment layout: per lane, K pattern is {0..7,16..23}
// (lanes 0-15) or {8..15,24..31} (lanes 16-31).
__device__ __forceinline__ v16h frag16(const _Float16* p0, const _Float16* p1) {
    v8h lo = *(const v8h*)p0;
    v8h hi = *(const v8h*)p1;
    return __builtin_shufflevector(lo, hi, 0,1,2,3,4,5,6,7,8,9,10,11,12,13,14,15);
}

__device__ __forceinline__ v8f wmma_f16(v16h a, v16h b, v8f c) {
    return __builtin_amdgcn_wmma_f32_16x16x32_f16(false, a, false, b, (short)0, c, false, false);
}

// ------------------------------------------------------------------ TDM
// Issue a Tensor-Data-Mover load of a 64(row) x 64(col) f16 tile of W
// (row stride DIM elements) into LDS at lds_byte_addr. D# built per the
// CDNA5 ISA bit layout (group0: count/lds/global/type; group1: data_size,
// dims, tile dims, stride; groups 2/3 zero for a 2D tensor). This toolchain
// exposes the 6-arg builtin (g0, g1, g2, g3, g4, cpol).
__device__ __forceinline__ void tdm_load_w_tile(const _Float16* gsrc, unsigned int lds_byte_addr) {
    const unsigned long long ga = (unsigned long long)(uintptr_t)gsrc;

    u32x4 g0;
    g0[0] = 1u;                                        // count=1, user mode
    g0[1] = lds_byte_addr;                             // lds_addr
    g0[2] = (unsigned int)(ga & 0xffffffffu);          // global_addr[31:0]
    g0[3] = (unsigned int)((ga >> 32) & 0x1ffffffu)    // global_addr[56:32]
          | (2u << 30);                                // type = 2 ("image")

    const unsigned int td0 = DIM;   // tensor_dim0 (elements per row)
    const unsigned int td1 = 64u;   // tensor_dim1 (rows available)
    const unsigned int tl0 = 64u;   // tile_dim0 (cols)
    const unsigned int tl1 = 64u;   // tile_dim1 (rows)
    i32x8 g1;
    g1[0] = (int)(1u << 16);                           // workgroup_mask=0, data_size=1 (2B)
    g1[1] = (int)((td0 & 0xffffu) << 16);              // atomic_barrier_addr=0 | dim0[15:0]
    g1[2] = (int)((td0 >> 16) | ((td1 & 0xffffu) << 16)); // dim0[31:16] | dim1[15:0]
    g1[3] = (int)((td1 >> 16) | (tl0 << 16));          // dim1[31:16] | tile_dim0
    g1[4] = (int)(tl1);                                // tile_dim1 | tile_dim2=0
    g1[5] = (int)DIM;                                  // tensor_dim0_stride[31:0]
    g1[6] = 0;                                         // stride[47:32] | dim1_stride lo (unused, 2D)
    g1[7] = 0;
    i32x4 g2 = (i32x4){0, 0, 0, 0};
    i32x4 g3 = (i32x4){0, 0, 0, 0};
    i32x8 g4 = (i32x8){0, 0, 0, 0, 0, 0, 0, 0};
    __builtin_amdgcn_tensor_load_to_lds(g0, g1, g2, g3, g4, 0);
}

// ---------------------------------------------------------------- convert
__global__ __launch_bounds__(256) void cvt_f32_f16(const float* __restrict__ in,
                                                   _Float16* __restrict__ out, int n) {
    int i = blockIdx.x * blockDim.x + threadIdx.x;
    if (i < n) out[i] = (_Float16)in[i];
}

// ---------------------------------------------------------------- GEMM
// Y = X @ W^T + bias.  X:[S_LEN,DIM] f16, W:[DIM,DIM] f16 row-major.
// mode 0: Y[s*DIM + o] (f32)     mode 1: Y[((o>>6)*S_LEN + s)*HD + (o&63)]
// grid: (S_LEN/64, DIM/64), block 128 (4 waves x 16 rows).
// W tile (shared by all 4 waves) is staged through LDS by the TDM,
// double-buffered, synchronized via TENSORcnt + workgroup barrier.
__global__ __launch_bounds__(128)
void gemm16(const _Float16* __restrict__ X, const _Float16* __restrict__ W,
            const float* __restrict__ bias, float* __restrict__ Y, int mode) {
    __shared__ _Float16 Wlds[2][64 * 64];   // 2 x 8KB stages

    const int lane = threadIdx.x & 31;
    const int wave = threadIdx.x >> 5;
    const int r    = lane & 15;
    const int hw   = lane >> 4;
    const int s0   = blockIdx.x * 64 + wave * 16;
    const int o0   = blockIdx.y * 64;
    const int NKT  = DIM / 64;              // 16 k-tiles of 64

    v8f acc[4] = {v8f{}, v8f{}, v8f{}, v8f{}};
    const _Float16* xrow = X + (s0 + r) * DIM;
    const unsigned int lds_base = (unsigned int)(uintptr_t)(void*)&Wlds[0][0];

    if (wave == 0) tdm_load_w_tile(W + o0 * DIM, lds_base);

    for (int kt = 0; kt < NKT; ++kt) {
        if (wave == 0) {
            if (kt + 1 < NKT) {
                tdm_load_w_tile(W + o0 * DIM + (kt + 1) * 64,
                                lds_base + ((kt + 1) & 1) * (64 * 64 * 2));
                __builtin_amdgcn_s_wait_tensorcnt(1);
            } else {
                __builtin_amdgcn_s_wait_tensorcnt(0);
            }
        }
        __syncthreads();                       // W stage (kt&1) visible to all waves

        const _Float16* wl = &Wlds[kt & 1][0];
        const int kg = kt * 64;
        __builtin_prefetch(xrow + kg + 256, 0, 3);   // global_prefetch_b8 (X stream)
#pragma unroll
        for (int kk = 0; kk < 2; ++kk) {
            const int k0 = kk * 32;
            v16h a = frag16(xrow + kg + k0 + hw * 8, xrow + kg + k0 + 16 + hw * 8);
            v16h b[4];
#pragma unroll
            for (int j = 0; j < 4; ++j) {
                const _Float16* wrow = wl + (j * 16 + r) * 64 + k0;
                b[j] = frag16(wrow + hw * 8, wrow + 16 + hw * 8);   // ds_load_b128
            }
#pragma unroll
            for (int j = 0; j < 4; ++j) acc[j] = wmma_f16(a, b[j], acc[j]);
        }
        __syncthreads();                       // done reading this stage
    }

#pragma unroll
    for (int j = 0; j < 4; ++j) {
        const int o = o0 + j * 16 + r;         // N = lane&15
        const float bv = bias[o];
#pragma unroll
        for (int rr = 0; rr < 8; ++rr) {
            const int s = s0 + rr + hw * 8;    // M = rr + 8*hw
            const float v = acc[j][rr] + bv;
            if (mode == 0) {
                Y[s * DIM + o] = v;
            } else {
                const int h = o >> 6, d = o & 63;
                Y[(h * S_LEN + s) * HD + d] = v;
            }
        }
    }
}

// ------------------------------------------------- RoPE + entangle + phase
__global__ __launch_bounds__(256)
void rope_mix_phase(const float* __restrict__ Qr, const float* __restrict__ Qi,
                    const float* __restrict__ Kr, const float* __restrict__ Ki,
                    const float* __restrict__ Vr, const float* __restrict__ Vi,
                    const float* __restrict__ phase, const float* __restrict__ ent,
                    const float* __restrict__ freqs,
                    _Float16* __restrict__ oQr, _Float16* __restrict__ oQi,
                    _Float16* __restrict__ oQs, _Float16* __restrict__ oKr,
                    _Float16* __restrict__ oKi, _Float16* __restrict__ oKs,
                    _Float16* __restrict__ oVr, _Float16* __restrict__ oVi) {
    const int idx = blockIdx.x * blockDim.x + threadIdx.x;   // over S_LEN*HD
    const int s = idx >> 6, d = idx & 63;
    const bool rot = d < ROT;
    float c = 1.f, sn = 0.f;
    if (rot) {
        const float ang = (float)s * freqs[d >> 1];
        c = cosf(ang); sn = sinf(ang);
    }
    const int d0 = d & ~1, d1 = d | 1;
    const bool odd = (d & 1) != 0;

    float qr[HEADS], qi[HEADS], kr[HEADS], ki[HEADS];
#pragma unroll
    for (int h = 0; h < HEADS; ++h) {
        const int base = (h * S_LEN + s) * HD;
        if (rot) {
            float x1, x2;
            x1 = Qr[base + d0]; x2 = Qr[base + d1];
            qr[h] = odd ? (x1 * sn + x2 * c) : (x1 * c - x2 * sn);
            x1 = Qi[base + d0]; x2 = Qi[base + d1];
            qi[h] = odd ? (x1 * sn + x2 * c) : (x1 * c - x2 * sn);
            x1 = Kr[base + d0]; x2 = Kr[base + d1];
            kr[h] = odd ? (x1 * sn + x2 * c) : (x1 * c - x2 * sn);
            x1 = Ki[base + d0]; x2 = Ki[base + d1];
            ki[h] = odd ? (x1 * sn + x2 * c) : (x1 * c - x2 * sn);
        } else {
            qr[h] = Qr[base + d]; qi[h] = Qi[base + d];
            kr[h] = Kr[base + d]; ki[h] = Ki[base + d];
        }
    }

#pragma unroll
    for (int hx = 0; hx < HEADS; ++hx) {
        float eqr = 0.f, eqi = 0.f, ekr = 0.f, eki = 0.f;
#pragma unroll
        for (int h = 0; h < HEADS; ++h) {
            const float e = ent[h * HEADS + hx];
            eqr += qr[h] * e; eqi += qi[h] * e;
            ekr += kr[h] * e; eki += ki[h] * e;
        }
        const float ph = phase[hx * HD + d];
        const float pc = cosf(ph), ps = sinf(ph);
        const float nqr = eqr * pc - eqi * ps, nqi = eqr * ps + eqi * pc;
        const float nkr = ekr * pc - eki * ps, nki = ekr * ps + eki * pc;
        const int o = (hx * S_LEN + s) * HD + d;
        oQr[o] = (_Float16)nqr; oQi[o] = (_Float16)nqi; oQs[o] = (_Float16)(nqr + nqi);
        oKr[o] = (_Float16)nkr; oKi[o] = (_Float16)nki; oKs[o] = (_Float16)(nkr + nki);
        const int vt = (hx * HD + d) * S_LEN + s;
        oVr[vt] = (_Float16)Vr[o];
        oVi[vt] = (_Float16)Vi[o];
    }
}

// ---------------------------------------------------------------- attention
// grid: (S_LEN/64, HEADS), block 128 (4 waves, 16 query rows each).
__global__ __launch_bounds__(128)
void attn(const _Float16* __restrict__ Qr, const _Float16* __restrict__ Qi,
          const _Float16* __restrict__ Qs, const _Float16* __restrict__ Kr,
          const _Float16* __restrict__ Ki, const _Float16* __restrict__ Ks,
          const _Float16* __restrict__ Vtr, const _Float16* __restrict__ Vti,
          const float* __restrict__ p_strength, const float* __restrict__ p_temp,
          const float* __restrict__ p_ceps,
          _Float16* __restrict__ Or16, _Float16* __restrict__ Oi16) {
    __shared__ _Float16 Plds[4][16 * 32];    // per-wave prob tile (no WG barriers)

    const int lane = threadIdx.x & 31;
    const int wave = threadIdx.x >> 5;
    const int r    = lane & 15;
    const int hw   = lane >> 4;
    const int h    = blockIdx.y;
    const int s0   = blockIdx.x * 64 + wave * 16;

    const float eps   = 0.03f / (1.f + expf(-p_ceps[0]));
    float temp        = expf(p_temp[0]);
    temp              = temp < 0.1f ? 0.1f : temp;
    const float coef  = (1.f / (1.f + expf(-p_strength[0]))) / temp;
    const float scale = 0.125f;              // 1/sqrt(64)

    const _Float16* qrrow = Qr + (h * S_LEN + s0 + r) * HD;
    const _Float16* qirow = Qi + (h * S_LEN + s0 + r) * HD;
    const _Float16* qsrow = Qs + (h * S_LEN + s0 + r) * HD;
    v16h qA[2], qiA[2], qsA[2];
#pragma unroll
    for (int dh = 0; dh < 2; ++dh) {
        qA[dh]  = frag16(qrrow + dh * 32 + hw * 8, qrrow + dh * 32 + 16 + hw * 8);
        qiA[dh] = frag16(qirow + dh * 32 + hw * 8, qirow + dh * 32 + 16 + hw * 8);
        qsA[dh] = frag16(qsrow + dh * 32 + hw * 8, qsrow + dh * 32 + 16 + hw * 8);
    }

    v8f outr[4] = {v8f{}, v8f{}, v8f{}, v8f{}};
    v8f outi[4] = {v8f{}, v8f{}, v8f{}, v8f{}};
    float mrun[8], lrun[8];
#pragma unroll
    for (int rr = 0; rr < 8; ++rr) { mrun[rr] = -INFINITY; lrun[rr] = 0.f; }

    for (int t0 = 0; t0 <= s0 + 15; t0 += 32) {
        // --- batch-load all 12 K B-fragments, then 12 WMMAs ---
        v16h kb1[2][2], kb2[2][2], kb3[2][2];
#pragma unroll
        for (int nt = 0; nt < 2; ++nt) {
            const int trow = (h * S_LEN + t0 + nt * 16 + r) * HD;
#pragma unroll
            for (int dh = 0; dh < 2; ++dh) {
                kb1[nt][dh] = frag16(Kr + trow + dh * 32 + hw * 8,
                                     Kr + trow + dh * 32 + 16 + hw * 8);
                kb2[nt][dh] = frag16(Ki + trow + dh * 32 + hw * 8,
                                     Ki + trow + dh * 32 + 16 + hw * 8);
                kb3[nt][dh] = frag16(Ks + trow + dh * 32 + hw * 8,
                                     Ks + trow + dh * 32 + 16 + hw * 8);
            }
        }
        v8f sc1[2] = {v8f{}, v8f{}}, sc2[2] = {v8f{}, v8f{}}, sc3[2] = {v8f{}, v8f{}};
#pragma unroll
        for (int nt = 0; nt < 2; ++nt)
#pragma unroll
            for (int dh = 0; dh < 2; ++dh) {
                sc1[nt] = wmma_f16(qA[dh],  kb1[nt][dh], sc1[nt]);
                sc2[nt] = wmma_f16(qiA[dh], kb2[nt][dh], sc2[nt]);
                sc3[nt] = wmma_f16(qsA[dh], kb3[nt][dh], sc3[nt]);
            }

        // --- magnitude + mask + online softmax, per row ---
#pragma unroll
        for (int rr = 0; rr < 8; ++rr) {
            const int srow = s0 + rr + hw * 8;
            float x[2];
#pragma unroll
            for (int nt = 0; nt < 2; ++nt) {
                const float k1 = sc1[nt][rr], k2 = sc2[nt][rr], k3 = sc3[nt][rr];
                const float ar0 = (k1 - k2) * scale;
                const float ai0 = (k3 - k1 - k2) * scale;
                const float ar = ar0 + eps * ai0;
                const float ai = ai0 - eps * ar0;
                const float mag = sqrtf(ar * ar + ai * ai + 1e-6f);
                const int t = t0 + nt * 16 + r;
                x[nt] = (t <= srow) ? mag * coef : -INFINITY;
            }
            float loc = fmaxf(x[0], x[1]);
#pragma unroll
            for (int m = 1; m < 16; m <<= 1) loc = fmaxf(loc, __shfl_xor(loc, m, 32));
            const float mo = mrun[rr];
            const float mn = fmaxf(mo, loc);
            const float corr = expf(mo - mn);
            mrun[rr] = mn;
            const float p0 = expf(x[0] - mn);
            const float p1 = expf(x[1] - mn);
            float psum = p0 + p1;
#pragma unroll
            for (int m = 1; m < 16; m <<= 1) psum += __shfl_xor(psum, m, 32);
            lrun[rr] = lrun[rr] * corr + psum;
#pragma unroll
            for (int j = 0; j < 4; ++j) { outr[j][rr] *= corr; outi[j][rr] *= corr; }
            const int prow = (rr + hw * 8) * 32 + r;
            Plds[wave][prow]      = (_Float16)p0;
            Plds[wave][prow + 16] = (_Float16)p1;
        }

        // --- P @ V accumulation ---
        const _Float16* pr = &Plds[wave][r * 32];
        const v16h pa = frag16(pr + hw * 8, pr + 16 + hw * 8);
#pragma unroll
        for (int j = 0; j < 4; ++j) {
            const int vrow = (h * HD + j * 16 + r) * S_LEN + t0;
            v16h vb = frag16(Vtr + vrow + hw * 8, Vtr + vrow + 16 + hw * 8);
            outr[j] = wmma_f16(pa, vb, outr[j]);
            vb = frag16(Vti + vrow + hw * 8, Vti + vrow + 16 + hw * 8);
            outi[j] = wmma_f16(pa, vb, outi[j]);
        }
    }

#pragma unroll
    for (int rr = 0; rr < 8; ++rr) {
        const int srow = s0 + rr + hw * 8;
        const float inv = 1.f / lrun[rr];
#pragma unroll
        for (int j = 0; j < 4; ++j) {
            const int o = srow * DIM + h * HD + j * 16 + r;
            Or16[o] = (_Float16)(outr[j][rr] * inv);
            Oi16[o] = (_Float16)(outi[j][rr] * inv);
        }
    }
}

// ---------------------------------------------------------------- launcher
extern "C" void kernel_launch(void* const* d_in, const int* in_sizes, int n_in,
                              void* d_out, int out_size, void* d_ws, size_t ws_size,
                              hipStream_t stream) {
    (void)in_sizes; (void)n_in; (void)out_size; (void)ws_size;

    const float* real  = (const float*)d_in[0];
    const float* imag  = (const float*)d_in[1];
    const float* Wf[8] = {(const float*)d_in[2],  (const float*)d_in[4],
                          (const float*)d_in[6],  (const float*)d_in[8],
                          (const float*)d_in[10], (const float*)d_in[12],
                          (const float*)d_in[14], (const float*)d_in[16]};
    const float* bq_r = (const float*)d_in[3];
    const float* bk_r = (const float*)d_in[5];
    const float* bv_r = (const float*)d_in[7];
    const float* bq_i = (const float*)d_in[9];
    const float* bk_i = (const float*)d_in[11];
    const float* bv_i = (const float*)d_in[13];
    const float* bo_r = (const float*)d_in[15];
    const float* bo_i = (const float*)d_in[17];
    const float* phase    = (const float*)d_in[18];
    const float* ent      = (const float*)d_in[19];
    const float* freqs    = (const float*)d_in[20];
    const float* strength = (const float*)d_in[21];
    const float* temp     = (const float*)d_in[22];
    const float* ceps     = (const float*)d_in[23];

    char* ws = (char*)d_ws;
    size_t off = 0;
    auto carve = [&](size_t bytes) { char* p = ws + off; off += bytes; return p; };
    const size_t ACT = (size_t)S_LEN * DIM;
    const size_t WEL = (size_t)DIM * DIM;
    const size_t PRJ = (size_t)HEADS * S_LEN * HD;

    _Float16* xr16 = (_Float16*)carve(ACT * 2);
    _Float16* xi16 = (_Float16*)carve(ACT * 2);
    _Float16* w16[8];
    for (int i = 0; i < 8; ++i) w16[i] = (_Float16*)carve(WEL * 2);
    float* Qrf = (float*)carve(PRJ * 4);
    float* Krf = (float*)carve(PRJ * 4);
    float* Vrf = (float*)carve(PRJ * 4);
    float* Qif = (float*)carve(PRJ * 4);
    float* Kif = (float*)carve(PRJ * 4);
    float* Vif = (float*)carve(PRJ * 4);
    _Float16* Qr16 = (_Float16*)carve(PRJ * 2);
    _Float16* Qi16 = (_Float16*)carve(PRJ * 2);
    _Float16* Qs16 = (_Float16*)carve(PRJ * 2);
    _Float16* Kr16 = (_Float16*)carve(PRJ * 2);
    _Float16* Ki16 = (_Float16*)carve(PRJ * 2);
    _Float16* Ks16 = (_Float16*)carve(PRJ * 2);
    _Float16* Vtr  = (_Float16*)carve(PRJ * 2);
    _Float16* Vti  = (_Float16*)carve(PRJ * 2);
    _Float16* Or16 = (_Float16*)carve(ACT * 2);
    _Float16* Oi16 = (_Float16*)carve(ACT * 2);

    cvt_f32_f16<<<(int)((ACT + 255) / 256), 256, 0, stream>>>(real, xr16, (int)ACT);
    cvt_f32_f16<<<(int)((ACT + 255) / 256), 256, 0, stream>>>(imag, xi16, (int)ACT);
    for (int i = 0; i < 8; ++i)
        cvt_f32_f16<<<(int)((WEL + 255) / 256), 256, 0, stream>>>(Wf[i], w16[i], (int)WEL);

    dim3 gg(S_LEN / 64, DIM / 64);
    gemm16<<<gg, 128, 0, stream>>>(xr16, w16[0], bq_r, Qrf, 1);
    gemm16<<<gg, 128, 0, stream>>>(xr16, w16[1], bk_r, Krf, 1);
    gemm16<<<gg, 128, 0, stream>>>(xr16, w16[2], bv_r, Vrf, 1);
    gemm16<<<gg, 128, 0, stream>>>(xi16, w16[3], bq_i, Qif, 1);
    gemm16<<<gg, 128, 0, stream>>>(xi16, w16[4], bk_i, Kif, 1);
    gemm16<<<gg, 128, 0, stream>>>(xi16, w16[5], bv_i, Vif, 1);

    rope_mix_phase<<<(S_LEN * HD) / 256, 256, 0, stream>>>(
        Qrf, Qif, Krf, Kif, Vrf, Vif, phase, ent, freqs,
        Qr16, Qi16, Qs16, Kr16, Ki16, Ks16, Vtr, Vti);

    dim3 ga(S_LEN / 64, HEADS);
    attn<<<ga, 128, 0, stream>>>(Qr16, Qi16, Qs16, Kr16, Ki16, Ks16, Vtr, Vti,
                                 strength, temp, ceps, Or16, Oi16);

    float* out_r = (float*)d_out;
    float* out_i = out_r + (size_t)S_LEN * DIM;
    gemm16<<<gg, 128, 0, stream>>>(Or16, w16[6], bo_r, out_r, 0);
    gemm16<<<gg, 128, 0, stream>>>(Oi16, w16[7], bo_i, out_i, 0);
}
